// VerifyChunk1_86397562126731
// MI455X (gfx1250) — matrix operand order repaired and checked by reading, not compiled, not run on hardware
//
#include <hip/hip_runtime.h>
#include <hip/hip_bf16.h>

// ---------------------------------------------------------------------------
// Model constants (from reference)
// ---------------------------------------------------------------------------
#define T_      128
#define D_      2048
#define NH_     8
#define KVH_    2
#define HD_     256
#define NREP_   4
#define DFF_    8192
#define PLE_    256
#define NLTOT_  30
#define STORE_  512
#define S_SL_   512
#define S_FULL_ 1024
#define EPS_    1e-6f

typedef __attribute__((ext_vector_type(16))) __bf16 v16bf;
typedef __attribute__((ext_vector_type(8)))  float  v8f;

// LDS offset of a __shared__ object: generic LDS addresses carry the LDS
// byte offset in the low 32 bits (ISA aperture layout), so truncation works.
__device__ __forceinline__ unsigned lds_off(const void* p) {
  return (unsigned)(unsigned long long)p;
}

// Async DMA: global (16B) -> LDS, tracked by ASYNCcnt.
__device__ __forceinline__ void async_copy_b128(unsigned lds_addr, const float* gaddr) {
  asm volatile("global_load_async_to_lds_b128 %0, %1, off"
               :: "v"(lds_addr), "v"(gaddr) : "memory");
}
__device__ __forceinline__ void wait_async0() {
  asm volatile("s_wait_asynccnt 0x0" ::: "memory");
}

// ---------------------------------------------------------------------------
// Generic batched GEMM:  C[z][m,n] = alpha * sum_k A[z][m,k] * B[z/bdiv][n,k]
//   A element (m,k) at m*lda + k            (row-major, k contiguous)
//   B element (n,k) at n*ldbn + k*ldbk      (NT: ldbk==1, NN: ldbn==1)
// fp32 in/out, bf16 WMMA (f32 accumulate) internally.
// Block tile 128x64x(BK=64), 8 waves, each wave 32x32 (2x2 WMMA 16x16x32 x2).
// Tiles staged f32 in double-buffered LDS via GLOBAL_LOAD_ASYNC_TO_LDS_B128;
// f32->bf16 conversion happens at fragment build (co-executes with WMMA).
// Fragments are gathered with explicit float4 LDS loads -> pure ds_load_b128.
// Requires: M multiple of 128, N multiple of 64, K multiple of 64.
// ---------------------------------------------------------------------------
__global__ __launch_bounds__(256) void gemm_wmma_kernel(
    const float* __restrict__ A, const float* __restrict__ B,
    float* __restrict__ C,
    int K, int lda, int ldbn, int ldbk, int ldc,
    long long sA, long long sB, long long sC, int bdiv,
    const float* __restrict__ alpha_ptr)
{
  __shared__ float Asf[2][128][64];   // 64 KB
  __shared__ float Bsf[2][64][64];    // 32 KB

  const int z = blockIdx.z;
  A += (long long)z * sA;
  B += (long long)(z / bdiv) * sB;
  C += (long long)z * sC;

  const int n0   = blockIdx.x * 64;
  const int m0   = blockIdx.y * 128;
  const int tid  = threadIdx.x;
  const int wave = tid >> 5;
  const int lane = tid & 31;
  const int wm   = (wave & 3) * 32;   // wave row base within block tile
  const int wn   = (wave >> 2) * 32;  // wave col base within block tile
  const int half = lane >> 4;
  const int l15  = lane & 15;

  v8f acc[2][2] = {};
  float brs[16];                      // B staging regs (scalar path only)

  // Issue the async DMA for one 64-deep K tile into LDS buffer `buf`.
  auto issue_tile = [&](int k0, int buf) {
    #pragma unroll
    for (int i = 0; i < 8; ++i) {     // A: 128x64 f32 = 2048 x b128
      const int idx = tid + i * 256;
      const int m   = idx >> 4;
      const int kk  = (idx & 15) * 4;
      async_copy_b128(lds_off(&Asf[buf][m][kk]),
                      A + (long long)(m0 + m) * lda + (k0 + kk));
    }
    if (ldbk == 1) {
      #pragma unroll
      for (int i = 0; i < 4; ++i) {   // B: 64x64 f32 = 1024 x b128
        const int idx = tid + i * 256;
        const int n   = idx >> 4;
        const int kk  = (idx & 15) * 4;
        async_copy_b128(lds_off(&Bsf[buf][n][kk]),
                        B + (long long)(n0 + n) * ldbn + (k0 + kk));
      }
    } else {
      #pragma unroll
      for (int i = 0; i < 16; ++i) {  // strided B: stage through registers
        const int idx = tid + i * 256;
        const int n   = idx >> 6;
        const int kk  = idx & 63;
        brs[i] = B[(long long)(n0 + n) * ldbn + (long long)(k0 + kk) * ldbk];
      }
    }
  };

  // Complete the staging of buffer `buf` (called after compute of prev tile).
  auto finish_tile = [&](int buf) {
    if (ldbk != 1) {
      #pragma unroll
      for (int i = 0; i < 16; ++i) {
        const int idx = tid + i * 256;
        Bsf[buf][idx >> 6][idx & 63] = brs[i];
      }
    }
    wait_async0();
    __syncthreads();
  };

  const int nk = K >> 6;              // number of 64-deep K stages

  issue_tile(0, 0);
  finish_tile(0);

  for (int t = 0; t < nk; ++t) {
    const int cur = t & 1;
    // kick off DMA for the next tile into the other buffer (safe: its readers
    // finished before the barrier that ended iteration t-1)
    if (t + 1 < nk) issue_tile((t + 1) << 6, cur ^ 1);
    // L2 prefetch two tiles ahead (global_prefetch_b8)
    if (t + 2 < nk) {
      const int pm = tid >> 1;
      __builtin_prefetch(A + (long long)(m0 + pm) * lda + ((t + 2) << 6) + (tid & 1) * 32, 0, 0);
    }

    // ---- compute: two 32-deep WMMA sub-steps from LDS[cur] ----
    #pragma unroll
    for (int s = 0; s < 2; ++s) {
      const int ks = s * 32;
      v16bf afrag[2], bfrag[2];
      #pragma unroll
      for (int mi = 0; mi < 2; ++mi) {
        const int row = wm + mi * 16 + l15;
        const int kb  = ks + half * 16; // lanes0-15 K0..15; lanes16-31 K16..31
        const float4 a0 = *(const float4*)&Asf[cur][row][kb + 0];
        const float4 a1 = *(const float4*)&Asf[cur][row][kb + 4];
        const float4 a2 = *(const float4*)&Asf[cur][row][kb + 8];
        const float4 a3 = *(const float4*)&Asf[cur][row][kb + 12];
        afrag[mi] = (v16bf){
            (__bf16)a0.x, (__bf16)a0.y, (__bf16)a0.z, (__bf16)a0.w,
            (__bf16)a1.x, (__bf16)a1.y, (__bf16)a1.z, (__bf16)a1.w,
            (__bf16)a2.x, (__bf16)a2.y, (__bf16)a2.z, (__bf16)a2.w,
            (__bf16)a3.x, (__bf16)a3.y, (__bf16)a3.z, (__bf16)a3.w};
      }
      #pragma unroll
      for (int ni = 0; ni < 2; ++ni) {
        const int col = wn + ni * 16 + l15;
        const int kb  = ks + half * 16; // lanes0-15 K0..15; lanes16-31 K16..31
        const float4 b0 = *(const float4*)&Bsf[cur][col][kb + 0];
        const float4 b1 = *(const float4*)&Bsf[cur][col][kb + 4];
        const float4 b2 = *(const float4*)&Bsf[cur][col][kb + 8];
        const float4 b3 = *(const float4*)&Bsf[cur][col][kb + 12];
        bfrag[ni] = (v16bf){
            (__bf16)b0.x, (__bf16)b0.y, (__bf16)b0.z, (__bf16)b0.w,
            (__bf16)b1.x, (__bf16)b1.y, (__bf16)b1.z, (__bf16)b1.w,
            (__bf16)b2.x, (__bf16)b2.y, (__bf16)b2.z, (__bf16)b2.w,
            (__bf16)b3.x, (__bf16)b3.y, (__bf16)b3.z, (__bf16)b3.w};
      }
      #pragma unroll
      for (int mi = 0; mi < 2; ++mi)
        #pragma unroll
        for (int ni = 0; ni < 2; ++ni)
          acc[mi][ni] = __builtin_amdgcn_wmma_f32_16x16x32_bf16(
              false, afrag[mi], false, bfrag[ni],
              (short)0, acc[mi][ni], false, false);
    }

    if (t + 1 < nk) finish_tile(cur ^ 1);
  }

  const float alpha = alpha_ptr ? *alpha_ptr : 1.0f;
  // C/D layout: lanes0-15 N=lane M=r; lanes16-31 N=lane-16 M=r+8
  #pragma unroll
  for (int mi = 0; mi < 2; ++mi)
    #pragma unroll
    for (int ni = 0; ni < 2; ++ni) {
      const int col = n0 + wn + ni * 16 + l15;
      #pragma unroll
      for (int r = 0; r < 8; ++r) {
        const int row = m0 + wm + mi * 16 + half * 8 + r;
        C[(long long)row * ldc + col] = acc[mi][ni][r] * alpha;
      }
    }
}

// ---------------------------------------------------------------------------
// Block reduction helper (blockDim.x == 256)
// ---------------------------------------------------------------------------
__device__ inline float block_sum(float v, float* red) {
  const int tid = threadIdx.x;
  red[tid] = v; __syncthreads();
  for (int s = 128; s > 0; s >>= 1) {
    if (tid < s) red[tid] += red[tid + s];
    __syncthreads();
  }
  float r = red[0]; __syncthreads();
  return r;
}

// ---------------------------------------------------------------------------
// RMS norm + affine + residual + scale.
//   mode 0: weight (1+w), 1: no weight, 2: weight w
//   out[row,c] = scale * ( (resid? resid[row,c] : 0) + rms(in[row])*wmode )
// One block per row (grid.x = rows), contiguous rows of `cols`.
// ---------------------------------------------------------------------------
__global__ __launch_bounds__(256) void rms_affine_kernel(
    const float* __restrict__ in, const float* __restrict__ w,
    const float* __restrict__ resid, float* __restrict__ out,
    int cols, int mode, const float* __restrict__ scale_ptr)
{
  __shared__ float red[256];
  const long long row = blockIdx.x;
  const float* xi = in + row * cols;
  float ss = 0.f;
  for (int c = threadIdx.x; c < cols; c += 256) { float v = xi[c]; ss += v * v; }
  const float total = block_sum(ss, red);
  const float inv   = rsqrtf(total / (float)cols + EPS_);
  const float scale = scale_ptr ? *scale_ptr : 1.0f;
  for (int c = threadIdx.x; c < cols; c += 256) {
    const float ww = (mode == 1) ? 1.f : ((mode == 0) ? (1.f + w[c]) : w[c]);
    float r = xi[c] * inv * ww;
    if (resid) r += resid[row * cols + c];
    out[row * cols + c] = r * scale;
  }
}

// ---------------------------------------------------------------------------
// RoPE in place on x laid out (T, heads, HD); cos/sin are (T, HD).
// blockDim.x == HD_ == 256.
// ---------------------------------------------------------------------------
__global__ __launch_bounds__(256) void rope_kernel(
    float* __restrict__ x, const float* __restrict__ cosT,
    const float* __restrict__ sinT, int heads)
{
  const int t = blockIdx.x, h = blockIdx.y, d = threadIdx.x;
  const long long idx = ((long long)t * heads + h) * HD_ + d;
  const float a   = x[idx];
  const float rot = (d < HD_ / 2) ? -x[idx + HD_ / 2] : x[idx - HD_ / 2];
  __syncthreads();  // all reads before any write
  x[idx] = a * cosT[t * HD_ + d] + rot * sinT[t * HD_ + d];
}

// ---------------------------------------------------------------------------
// Build contiguous KV cache: out (KVH, Stot, HD) from
//   cache (KVH, Sprev, ldcache) cols[0:HD]  and  cur (T, KVH, HD).
// grid (Stot, KVH)
// ---------------------------------------------------------------------------
__global__ __launch_bounds__(256) void concat_cache_kernel(
    const float* __restrict__ cache, const float* __restrict__ cur,
    float* __restrict__ out, int Sprev, int ldcache)
{
  const int s = blockIdx.x, kv = blockIdx.y;
  const int Stot = Sprev + T_;
  for (int d = threadIdx.x; d < HD_; d += 256) {
    const float v = (s < Sprev)
        ? cache[((long long)kv * Sprev + s) * ldcache + d]
        : cur[((long long)(s - Sprev) * KVH_ + kv) * HD_ + d];
    out[((long long)kv * Stot + s) * HD_ + d] = v;
  }
}

// ---------------------------------------------------------------------------
// Masked softmax over scores (NH, T, S); mask is (T, S). grid (T, NH).
// ---------------------------------------------------------------------------
__global__ __launch_bounds__(256) void softmax_kernel(
    float* __restrict__ sc, const float* __restrict__ mask, int S)
{
  __shared__ float red[256];
  const int t = blockIdx.x, h = blockIdx.y;
  float* row = sc + ((long long)h * T_ + t) * S;
  const float* mrow = mask + (long long)t * S;

  float mx = -1e30f;
  for (int s = threadIdx.x; s < S; s += 256) mx = fmaxf(mx, row[s] + mrow[s]);
  red[threadIdx.x] = mx; __syncthreads();
  for (int st = 128; st > 0; st >>= 1) {
    if (threadIdx.x < st) red[threadIdx.x] = fmaxf(red[threadIdx.x], red[threadIdx.x + st]);
    __syncthreads();
  }
  mx = red[0]; __syncthreads();

  float sum = 0.f;
  for (int s = threadIdx.x; s < S; s += 256) {
    const float e = __expf(row[s] + mrow[s] - mx);
    row[s] = e; sum += e;
  }
  const float tot = block_sum(sum, red);
  const float r = 1.f / tot;
  for (int s = threadIdx.x; s < S; s += 256) row[s] *= r;
}

// ---------------------------------------------------------------------------
// out[row,c] = gelu_tanh(g[row,c]) * (u ? u[row*ldu + c] : 1)
// grid.x = rows
// ---------------------------------------------------------------------------
__global__ __launch_bounds__(256) void gelu_mul_kernel(
    const float* __restrict__ g, const float* __restrict__ u,
    float* __restrict__ out, int cols, int ldu)
{
  const long long row = blockIdx.x;
  for (int c = threadIdx.x; c < cols; c += 256) {
    const float x  = g[row * cols + c];
    const float tt = 0.7978845608028654f * (x + 0.044715f * x * x * x);
    const float ge = 0.5f * x * (1.f + tanhf(tt));
    const float uu = u ? u[row * ldu + c] : 1.f;
    out[row * cols + c] = ge * uu;
  }
}

// ---------------------------------------------------------------------------
// Write new cache slice: k (T, KVH, HD) -> out (KVH, T, store) zero-padded.
// grid (T, KVH)
// ---------------------------------------------------------------------------
__global__ __launch_bounds__(256) void write_cache_kernel(
    const float* __restrict__ k, float* __restrict__ out, int store)
{
  const int t = blockIdx.x, kv = blockIdx.y;
  for (int d = threadIdx.x; d < store; d += 256) {
    const float v = (d < HD_) ? k[((long long)t * KVH_ + kv) * HD_ + d] : 0.f;
    out[((long long)kv * T_ + t) * store + d] = v;
  }
}

// ---------------------------------------------------------------------------
// Host orchestration
// ---------------------------------------------------------------------------
extern "C" void kernel_launch(void* const* d_in, const int* in_sizes, int n_in,
                              void* d_out, int out_size, void* d_ws, size_t ws_size,
                              hipStream_t stream) {
  (void)in_sizes; (void)n_in; (void)out_size; (void)ws_size;

  const float* hidden    = (const float*)d_in[0];
  const float* mask_full = (const float*)d_in[1];
  const float* mask_sl   = (const float*)d_in[2];
  const float* raw       = (const float*)d_in[3];
  const float* cos_s     = (const float*)d_in[4];
  const float* sin_s     = (const float*)d_in[5];
  const float* cos_f     = (const float*)d_in[6];
  const float* sin_f     = (const float*)d_in[7];
  const float* Ksl_in    = (const float*)d_in[8];
  const float* Vsl_in    = (const float*)d_in[9];
  const float* Kfu_in    = (const float*)d_in[10];
  const float* Vfu_in    = (const float*)d_in[11];
  // params: 8 layers x 17 entries, then Wple, ple_norm_w, ple_proj_scale, ple_input_scale
  auto LP = [&](int li, int j) -> const float* { return (const float*)d_in[12 + li * 17 + j]; };
  const float* Wple            = (const float*)d_in[12 + 8 * 17 + 0];
  const float* ple_norm_w      = (const float*)d_in[12 + 8 * 17 + 1];
  const float* ple_proj_scale  = (const float*)d_in[12 + 8 * 17 + 2];
  const float* ple_input_scale = (const float*)d_in[12 + 8 * 17 + 3];

  // workspace layout (floats)
  float* ws = (float*)d_ws;
  float* xbuf   = ws;              ws += (long long)T_ * D_;          // 262144
  float* qbuf   = ws;              ws += (long long)T_ * NH_ * HD_;   // 262144
  float* kbuf   = ws;              ws += (long long)T_ * KVH_ * HD_;  // 65536
  float* vbuf   = ws;              ws += (long long)T_ * KVH_ * HD_;  // 65536
  float* Kcat   = ws;              ws += (long long)KVH_ * (S_FULL_ + T_) * HD_; // 589824
  float* Vcat   = ws;              ws += (long long)KVH_ * (S_FULL_ + T_) * HD_;
  float* scores = ws;              ws += (long long)NH_ * T_ * (S_FULL_ + T_);   // 1179648
  float* aobuf  = ws;              ws += (long long)T_ * NH_ * HD_;
  float* gbuf   = ws;              ws += (long long)T_ * DFF_;        // 1048576
  float* ubuf   = ws;              ws += (long long)T_ * DFF_;
  float* t3     = ws;              ws += (long long)T_ * D_;
  float* hbuf   = ws;              ws += (long long)T_ * D_;
  float* gplbuf = ws;              ws += (long long)T_ * PLE_;

  // output layout (floats)
  float* out      = (float*)d_out;
  float* out_h    = out;
  float* out_plc  = out_h   + (long long)T_ * D_;
  float* out_Ks   = out_plc + (long long)T_ * NLTOT_ * PLE_;
  float* out_Vs   = out_Ks  + (long long)7 * KVH_ * T_ * STORE_;
  float* out_Kf   = out_Vs  + (long long)7 * KVH_ * T_ * STORE_;
  float* out_Vf   = out_Kf  + (long long)KVH_ * T_ * HD_;

  auto gemm = [&](const float* A, const float* B, float* C, int M, int N, int K,
                  int lda, int ldbn, int ldbk, int ldc,
                  long long sA, long long sB, long long sC,
                  int bdiv, int batch, const float* alpha) {
    dim3 grid(N / 64, M / 128, batch);
    gemm_wmma_kernel<<<grid, 256, 0, stream>>>(A, B, C, K, lda, ldbn, ldbk, ldc,
                                               sA, sB, sC, bdiv, alpha);
  };

  // h := hidden_states
  hipMemcpyAsync(hbuf, hidden, (size_t)T_ * D_ * sizeof(float),
                 hipMemcpyDeviceToDevice, stream);

  // ---- per-layer-embedding combine (uses `scores` as temp, free here) ----
  // proj = hidden @ Wple^T * ple_proj_scale : (T, 7680)
  gemm(hidden, Wple, scores, T_, NLTOT_ * PLE_, D_,
       D_, D_, 1, NLTOT_ * PLE_, 0, 0, 0, 1, 1, ple_proj_scale);
  // combined = (rms_nw(proj blocks of 256) * ple_norm_w + raw) * ple_input_scale
  rms_affine_kernel<<<dim3(T_ * NLTOT_), 256, 0, stream>>>(
      scores, ple_norm_w, raw, out_plc, PLE_, /*mode=*/2, ple_input_scale);

  for (int li = 0; li < 8; ++li) {
    const bool  is_full = (li == 4);
    const int   Sprev   = is_full ? S_FULL_ : S_SL_;
    const int   Stot    = Sprev + T_;
    const int   idx     = is_full ? 0 : (li < 4 ? li : li - 1);
    const float* cosT   = is_full ? cos_f : cos_s;
    const float* sinT   = is_full ? sin_f : sin_s;
    const float* maskT  = is_full ? mask_full : mask_sl;
    const float* Kc     = is_full ? Kfu_in : Ksl_in + (long long)idx * KVH_ * S_SL_ * STORE_;
    const float* Vc     = is_full ? Vfu_in : Vsl_in + (long long)idx * KVH_ * S_SL_ * STORE_;

    // --- attention ---
    rms_affine_kernel<<<dim3(T_), 256, 0, stream>>>(hbuf, LP(li, 0), nullptr, xbuf, D_, 0, nullptr);
    gemm(xbuf, LP(li, 7), qbuf, T_, NH_ * HD_, D_, D_, D_, 1, NH_ * HD_, 0, 0, 0, 1, 1, nullptr);
    gemm(xbuf, LP(li, 8), kbuf, T_, KVH_ * HD_, D_, D_, D_, 1, KVH_ * HD_, 0, 0, 0, 1, 1, nullptr);
    gemm(xbuf, LP(li, 9), vbuf, T_, KVH_ * HD_, D_, D_, D_, 1, KVH_ * HD_, 0, 0, 0, 1, 1, nullptr);
    rms_affine_kernel<<<dim3(T_ * NH_),  256, 0, stream>>>(qbuf, LP(li, 5), nullptr, qbuf, HD_, 0, nullptr);
    rms_affine_kernel<<<dim3(T_ * KVH_), 256, 0, stream>>>(kbuf, LP(li, 6), nullptr, kbuf, HD_, 0, nullptr);
    rms_affine_kernel<<<dim3(T_ * KVH_), 256, 0, stream>>>(vbuf, nullptr,   nullptr, vbuf, HD_, 1, nullptr);
    rope_kernel<<<dim3(T_, NH_),  256, 0, stream>>>(qbuf, cosT, sinT, NH_);
    rope_kernel<<<dim3(T_, KVH_), 256, 0, stream>>>(kbuf, cosT, sinT, KVH_);
    concat_cache_kernel<<<dim3(Stot, KVH_), 256, 0, stream>>>(Kc, kbuf, Kcat, Sprev, STORE_);
    concat_cache_kernel<<<dim3(Stot, KVH_), 256, 0, stream>>>(Vc, vbuf, Vcat, Sprev, STORE_);
    // scores[h] = q[h] @ Kcat[h/4]^T : (T, Stot), batched over NH heads
    gemm(qbuf, Kcat, scores, T_, Stot, HD_,
         NH_ * HD_, HD_, 1, Stot,
         /*sA=*/HD_, /*sB=*/(long long)Stot * HD_, /*sC=*/(long long)T_ * Stot,
         NREP_, NH_, nullptr);
    softmax_kernel<<<dim3(T_, NH_), 256, 0, stream>>>(scores, maskT, Stot);
    // ao[h] = softmax @ Vcat[h/4] : (T, HD), write interleaved as (T, NH*HD)
    gemm(scores, Vcat, aobuf, T_, HD_, Stot,
         Stot, 1, HD_, NH_ * HD_,
         (long long)T_ * Stot, (long long)Stot * HD_, HD_,
         NREP_, NH_, nullptr);
    gemm(aobuf, LP(li, 10), t3, T_, D_, NH_ * HD_, NH_ * HD_, NH_ * HD_, 1, D_, 0, 0, 0, 1, 1, nullptr);
    rms_affine_kernel<<<dim3(T_), 256, 0, stream>>>(t3, LP(li, 1), hbuf, hbuf, D_, 0, nullptr);

    // --- feed-forward ---
    rms_affine_kernel<<<dim3(T_), 256, 0, stream>>>(hbuf, LP(li, 2), nullptr, xbuf, D_, 0, nullptr);
    gemm(xbuf, LP(li, 11), gbuf, T_, DFF_, D_, D_, D_, 1, DFF_, 0, 0, 0, 1, 1, nullptr);
    gemm(xbuf, LP(li, 12), ubuf, T_, DFF_, D_, D_, D_, 1, DFF_, 0, 0, 0, 1, 1, nullptr);
    gelu_mul_kernel<<<dim3(T_), 256, 0, stream>>>(gbuf, ubuf, gbuf, DFF_, DFF_);
    gemm(gbuf, LP(li, 13), t3, T_, D_, DFF_, DFF_, DFF_, 1, D_, 0, 0, 0, 1, 1, nullptr);
    rms_affine_kernel<<<dim3(T_), 256, 0, stream>>>(t3, LP(li, 3), hbuf, hbuf, D_, 0, nullptr);

    // --- per-layer embedding block ---
    gemm(hbuf, LP(li, 14), gplbuf, T_, PLE_, D_, D_, D_, 1, PLE_, 0, 0, 0, 1, 1, nullptr);
    gelu_mul_kernel<<<dim3(T_), 256, 0, stream>>>(
        gplbuf, out_plc + (long long)li * PLE_, gplbuf, PLE_, NLTOT_ * PLE_);
    gemm(gplbuf, LP(li, 15), t3, T_, D_, PLE_, PLE_, PLE_, 1, D_, 0, 0, 0, 1, 1, nullptr);
    rms_affine_kernel<<<dim3(T_), 256, 0, stream>>>(t3, LP(li, 4), hbuf, hbuf, D_, 0, LP(li, 16));

    // --- cache outputs ---
    if (is_full) {
      write_cache_kernel<<<dim3(T_, KVH_), 256, 0, stream>>>(kbuf, out_Kf, HD_);
      write_cache_kernel<<<dim3(T_, KVH_), 256, 0, stream>>>(vbuf, out_Vf, HD_);
    } else {
      write_cache_kernel<<<dim3(T_, KVH_), 256, 0, stream>>>(
          kbuf, out_Ks + (long long)idx * KVH_ * T_ * STORE_, STORE_);
      write_cache_kernel<<<dim3(T_, KVH_), 256, 0, stream>>>(
          vbuf, out_Vs + (long long)idx * KVH_ * T_ * STORE_, STORE_);
    }
  }

  // final hidden state
  hipMemcpyAsync(out_h, hbuf, (size_t)T_ * D_ * sizeof(float),
                 hipMemcpyDeviceToDevice, stream);
}
